// ChunkedMultiHeadCardPassingLayer_33139967656626
// MI455X (gfx1250) — matrix operand order repaired and checked by reading, not compiled, not run on hardware
//
#include <hip/hip_runtime.h>
#include <hip/hip_bf16.h>

// ---------------------------------------------------------------- types
typedef __attribute__((ext_vector_type(16))) __bf16 v16bf;
typedef __attribute__((ext_vector_type(8)))  __bf16 v8bf;
typedef __attribute__((ext_vector_type(8)))  float  v8f;
typedef __attribute__((ext_vector_type(4)))  unsigned int v4u;
typedef __attribute__((ext_vector_type(8)))  int v8i_t;
typedef __attribute__((ext_vector_type(4)))  int v4i_t;

#if defined(__HIP_DEVICE_COMPILE__) && __has_builtin(__builtin_amdgcn_tensor_load_to_lds)
#define HAVE_TDM 1
#else
#define HAVE_TDM 0
#endif

// ---------------------------------------------------------------- shapes
constexpr int Bn  = 4;
constexpr int Tn  = 4096;
constexpr int Cn  = 1024;
constexpr int Hn  = 16;
constexpr int CSn = 128;
constexpr int Dn  = 64;
constexpr int NCH = Tn / CSn;      // 32
constexpr int Mn  = Bn * Tn;       // 16384
constexpr float EPSf = 1e-5f;

// ---------------------------------------------------------------- frag loaders
// A (16x32 bf16, row-major, row = lane&15):
//   elems 0..7  : K = half*8 + i
//   elems 8..15 : K = 16 + half*8 + (i-8)
__device__ __forceinline__ v16bf frag_a_rowmajor(const __bf16* rowp, int kk, int half) {
  v8bf a = *(const v8bf*)(rowp + kk + half * 8);
  v8bf b = *(const v8bf*)(rowp + kk + 16 + half * 8);
  v16bf r;
#pragma unroll
  for (int i = 0; i < 8; ++i) { r[i] = a[i]; r[i + 8] = b[i]; }
  return r;
}

// B (32x16 bf16) from a TRANSPOSED weight row (col n contiguous over K):
//   col = lane&15, elems 0..7: K = half*8 + i, elems 8..15: K = 16 + half*8 + (i-8)
__device__ __forceinline__ v16bf frag_b_t(const __bf16* colp, int kk, int half) {
  v8bf a = *(const v8bf*)(colp + kk + half * 8);
  v8bf b = *(const v8bf*)(colp + kk + 16 + half * 8);
  v16bf r;
#pragma unroll
  for (int i = 0; i < 8; ++i) { r[i] = a[i]; r[i + 8] = b[i]; }
  return r;
}

__device__ __forceinline__ v8f wmma_bf16(v16bf a, v16bf b, v8f c) {
  return __builtin_amdgcn_wmma_f32_16x16x32_bf16(false, a, false, b, (short)0, c,
                                                 false, false);
}

__device__ __forceinline__ float wave_sum32(float v) {
#pragma unroll
  for (int m = 16; m > 0; m >>= 1) v += __shfl_xor(v, m, 32);
  return v;
}

#if HAVE_TDM
// ---------------------------------------------------------------- TDM (Tensor Data Mover)
// Build a D# (ISA 8.3-8.6) for a 2D tile load: tile_d1 rows of tile_d0
// elements, global row stride stride0 (elements), element size 2^ds_code bytes.
// Optional LDS padding inserted every (2<<pad_int_code) DWORDs of (pad_amt_code+1) DWORDs.
__device__ __forceinline__ void tdm_load_2d(const void* gsrc, unsigned int lds_byte_addr,
                                            unsigned int tile_d0, unsigned int tile_d1,
                                            unsigned int tensor_d0, unsigned int tensor_d1,
                                            unsigned long long stride0,
                                            unsigned int ds_code,       // 0:1B 1:2B 2:4B
                                            unsigned int pad_enable,
                                            unsigned int pad_int_code,
                                            unsigned int pad_amt_code) {
  unsigned long long ga = (unsigned long long)gsrc;
  v4u g0;
  g0[0] = 1u;                                       // count=1, user descriptor
  g0[1] = lds_byte_addr;                            // lds_addr
  g0[2] = (unsigned int)ga;                         // global_addr[31:0]
  g0[3] = (unsigned int)(ga >> 32) | (2u << 30);    // global_addr[56:32] | type=2
  v8i_t g1;
  g1[0] = (int)((ds_code << 16) | (pad_enable << 20) |
                (pad_int_code << 22) | (pad_amt_code << 25)); // mask=0,data_size,pad
  g1[1] = (int)((tensor_d0 & 0xffffu) << 16);                 // tensor_dim0[15:0]
  g1[2] = (int)((tensor_d0 >> 16) | ((tensor_d1 & 0xffffu) << 16));
  g1[3] = (int)((tensor_d1 >> 16) | ((tile_d0 & 0xffffu) << 16)); // tile_dim0
  g1[4] = (int)(tile_d1 & 0xffffu);                           // tile_dim1 (tile_dim2=0)
  g1[5] = (int)(unsigned int)(stride0 & 0xffffffffull);       // dim0_stride[31:0]
  g1[6] = (int)(unsigned int)((stride0 >> 32) & 0xffffull);   // dim0_stride[47:32]
  g1[7] = 0;
  v4i_t z4 = {0, 0, 0, 0};
  v8i_t z8 = {0, 0, 0, 0, 0, 0, 0, 0};
  __builtin_amdgcn_tensor_load_to_lds(g0, g1, z4, z4, z8, 0);
}
__device__ __forceinline__ unsigned int lds_addr32(const void* p) {
  return (unsigned int)(unsigned long long)p;  // generic shared ptr: low 32 bits = LDS offset
}
#endif

// ---------------------------------------------------------------- precision prep
__global__ void k_cvt_bf16(const float* __restrict__ in, __bf16* __restrict__ out, int n) {
  for (int i = blockIdx.x * blockDim.x + threadIdx.x; i < n; i += gridDim.x * blockDim.x)
    out[i] = (__bf16)in[i];
}

// out[n*K + k] = in[k*N + n]   (transpose + convert)
__global__ void k_transpose_bf16(const float* __restrict__ in, __bf16* __restrict__ out,
                                 int K, int N) {
  int total = K * N;
  for (int i = blockIdx.x * blockDim.x + threadIdx.x; i < total; i += gridDim.x * blockDim.x) {
    int k = i % K, n = i / K;
    out[i] = (__bf16)in[k * N + n];
  }
}

// ---------------------------------------------------------------- K1: gated = sigmoid(x@gW+gb) * (x@mW+mb)
__global__ __launch_bounds__(256) void k_gated_gemm(
    const __bf16* __restrict__ xb, const __bf16* __restrict__ mWt,
    const __bf16* __restrict__ gWt, const float* __restrict__ mb,
    const float* __restrict__ gb, float* __restrict__ gated) {
  const int tid = threadIdx.x, lane = tid & 31, w = tid >> 5, half = lane >> 4;
  const int m0 = blockIdx.x * 64 + (w >> 1) * 16;
  const int n0 = blockIdx.y * 64 + (w & 1) * 32;

  const __bf16* arow = xb  + (size_t)(m0 + (lane & 15)) * Cn;
  const __bf16* mr0  = mWt + (size_t)(n0 +      (lane & 15)) * Cn;
  const __bf16* mr1  = mWt + (size_t)(n0 + 16 + (lane & 15)) * Cn;
  const __bf16* gr0  = gWt + (size_t)(n0 +      (lane & 15)) * Cn;
  const __bf16* gr1  = gWt + (size_t)(n0 + 16 + (lane & 15)) * Cn;

  v8f am0 = {}, am1 = {}, ag0 = {}, ag1 = {};
  for (int kk = 0; kk < Cn; kk += 32) {
    v16bf a = frag_a_rowmajor(arow, kk, half);
    am0 = wmma_bf16(a, frag_b_t(mr0, kk, half), am0);
    am1 = wmma_bf16(a, frag_b_t(mr1, kk, half), am1);
    ag0 = wmma_bf16(a, frag_b_t(gr0, kk, half), ag0);
    ag1 = wmma_bf16(a, frag_b_t(gr1, kk, half), ag1);
  }
#pragma unroll
  for (int nt = 0; nt < 2; ++nt) {
    const v8f& am = nt ? am1 : am0;
    const v8f& ag = nt ? ag1 : ag0;
    int col = n0 + nt * 16 + (lane & 15);
    float mbias = mb[col], gbias = gb[col];
#pragma unroll
    for (int i = 0; i < 8; ++i) {
      int row = m0 + i + 8 * half;
      float mv = am[i] + mbias;
      float gz = ag[i] + gbias;
      float gs = 1.0f / (1.0f + __expf(-gz));
      gated[(size_t)row * Cn + col] = mv * gs;
    }
  }
}

// ---------------------------------------------------------------- K2a: per-chunk sums
__global__ void k_chunk_sums(const float* __restrict__ gated, float* __restrict__ csum) {
  int g = blockIdx.x;                       // (b*H + h)*NCH + ch
  int ch = g % NCH, bh = g / NCH;
  int h = bh % Hn, b = bh / Hn;
  int d = threadIdx.x;                      // 64 threads
  const float* p = gated + ((size_t)b * Tn + (size_t)ch * CSn) * Cn + h * Dn + d;
  float s = 0.f;
  for (int t = 0; t < CSn; ++t) s += p[(size_t)t * Cn];
  csum[(size_t)g * Dn + d] = s;
}

// ---------------------------------------------------------------- K2b: carry scan + LN(D)
__global__ void k_carry_ln(const float* __restrict__ csum, const float* __restrict__ cg,
                           const float* __restrict__ cb, float* __restrict__ ncarry) {
  int bh = blockIdx.x;
  int l = threadIdx.x;                      // 32 lanes
  float c0 = 0.f, c1 = 0.f;
  float g0 = cg[l], g1 = cg[l + 32], b0 = cb[l], b1v = cb[l + 32];
  for (int ch = 0; ch < NCH; ++ch) {
    float mu = wave_sum32(c0 + c1) * (1.0f / 64.0f);
    float q0 = c0 - mu, q1 = c1 - mu;
    float var = wave_sum32(q0 * q0 + q1 * q1) * (1.0f / 64.0f);
    float rs = rsqrtf(var + EPSf);
    size_t idx = ((size_t)bh * NCH + ch) * Dn;
    ncarry[idx + l]      = q0 * rs * g0 + b0;
    ncarry[idx + l + 32] = q1 * rs * g1 + b1v;
    c0 += csum[idx + l];
    c1 += csum[idx + l + 32];
  }
}

// ---------------------------------------------------------------- K3: chunk scan + card LN + fused MLP (WMMA)
__global__ __launch_bounds__(256) void k_scan_mlp(
    const float* __restrict__ gated, const __bf16* __restrict__ xb,
    const float* __restrict__ ncarry, const float* __restrict__ card_g,
    const float* __restrict__ card_b, const __bf16* __restrict__ w1t_g,
    const float* __restrict__ b1, const float* __restrict__ alpha_p,
    const __bf16* __restrict__ w2t_g, const float* __restrict__ b2,
    __bf16* __restrict__ mlp_out) {
  __shared__ __attribute__((aligned(16))) float lds_g[CSn][Dn];      // 32 KB
  __shared__ float seg_tot[4][Dn];                                   // 1 KB
  __shared__ float ncar[Dn];                                         // 256 B
  __shared__ __attribute__((aligned(16))) __bf16 comb[CSn][2 * Dn];  // 32 KB
  __shared__ __attribute__((aligned(16))) __bf16 w1t[2 * Dn][2 * Dn];// 32 KB
  __shared__ __attribute__((aligned(16))) __bf16 w2t[Dn][2 * Dn];    // 16 KB
  __shared__ __attribute__((aligned(16))) __bf16 hbuf[CSn][2 * Dn];  // 32 KB

  const int tid = threadIdx.x, lane = tid & 31, w = tid >> 5, half = lane >> 4;
  const int g = blockIdx.x;
  const int ch = g % NCH, bh = g / NCH, h = bh % Hn, b = bh / Hn;
  const size_t row0 = ((size_t)b * Tn + (size_t)ch * CSn) * Cn + (size_t)h * Dn;

#if HAVE_TDM
  // Wave-level DMA staging through the Tensor Data Mover (EXEC-independent,
  // guarded by wave-uniform branches; each issuing wave drains its TENSORcnt).
  if (w == 0) {
    // gated chunk: 128 rows x 64 f32, global row stride 1024 f32 -> lds_g (dense)
    tdm_load_2d(gated + row0, lds_addr32(&lds_g[0][0]),
                Dn, CSn, Cn, CSn, Cn, /*ds=*/2, 0, 0, 0);
    __builtin_amdgcn_s_wait_tensorcnt(0);
  } else if (w == 1) {
    // x chunk: 128 rows x 64 bf16 -> comb[:,0:64], padding 32 DWORDs after every
    // 32 DWORDs to skip the cards half of each 256B combined row.
    tdm_load_2d(xb + row0, lds_addr32(&comb[0][0]),
                Dn, CSn, Cn, CSn, Cn, /*ds=*/1, 1, /*int:2<<4=32dw*/4, /*amt:31+1=32dw*/31);
    __builtin_amdgcn_s_wait_tensorcnt(0);
  } else if (w == 2) {
    // W1^T: flat 16384 bf16
    tdm_load_2d(w1t_g, lds_addr32(&w1t[0][0]),
                2 * Dn * 2 * Dn, 1, 2 * Dn * 2 * Dn, 1, 2 * Dn * 2 * Dn, 1, 0, 0, 0);
    __builtin_amdgcn_s_wait_tensorcnt(0);
  } else if (w == 3) {
    // W2^T: flat 8192 bf16
    tdm_load_2d(w2t_g, lds_addr32(&w2t[0][0]),
                Dn * 2 * Dn, 1, Dn * 2 * Dn, 1, Dn * 2 * Dn, 1, 0, 0, 0);
    __builtin_amdgcn_s_wait_tensorcnt(0);
  }
#else
  for (int i = tid; i < CSn * Dn; i += 256) {
    int t = i >> 6, d = i & 63;
    lds_g[t][d] = gated[row0 + (size_t)t * Cn + d];
    comb[t][d]  = xb[row0 + (size_t)t * Cn + d];
  }
  for (int i = tid; i < 2 * Dn * 2 * Dn; i += 256) (&w1t[0][0])[i] = w1t_g[i];
  for (int i = tid; i < Dn * 2 * Dn;     i += 256) (&w2t[0][0])[i] = w2t_g[i];
#endif
  if (tid < Dn) ncar[tid] = ncarry[((size_t)bh * NCH + ch) * Dn + tid];
  __syncthreads();

  // exclusive cumsum along tokens, per column d (4 segments x 64 columns)
  {
    int seg = tid >> 6, d = tid & 63;
    float run = 0.f;
    for (int j = 0; j < 32; ++j) {
      int t = seg * 32 + j;
      float v = lds_g[t][d];
      lds_g[t][d] = run;
      run += v;
    }
    seg_tot[seg][d] = run;
    __syncthreads();
    float off = 0.f;
    for (int s = 0; s < seg; ++s) off += seg_tot[s][d];
    for (int j = 0; j < 32; ++j) lds_g[seg * 32 + j][d] += off;
  }
  __syncthreads();

  // cards = LN_D(ncarry + exclusive_cumsum) -> comb[:,64:128]  (wave per 16 tokens)
  {
    float cg0 = card_g[lane], cg1 = card_g[lane + 32];
    float cb0 = card_b[lane], cb1 = card_b[lane + 32];
    float n0v = ncar[lane],   n1v = ncar[lane + 32];
    for (int t = w * 16; t < w * 16 + 16; ++t) {
      float e0 = n0v + lds_g[t][lane];
      float e1 = n1v + lds_g[t][lane + 32];
      float mu = wave_sum32(e0 + e1) * (1.0f / 64.0f);
      float q0 = e0 - mu, q1 = e1 - mu;
      float var = wave_sum32(q0 * q0 + q1 * q1) * (1.0f / 64.0f);
      float rs = rsqrtf(var + EPSf);
      comb[t][Dn + lane]      = (__bf16)(q0 * rs * cg0 + cb0);
      comb[t][Dn + lane + 32] = (__bf16)(q1 * rs * cg1 + cb1);
    }
  }
  __syncthreads();

  const float alpha = alpha_p[0];
  const int m0 = w * 16;

  // GEMM1: h(128x128) = combined(128x128) @ W1
  {
    v8f acc[8] = {};
    const __bf16* arow = &comb[m0 + (lane & 15)][0];
#pragma unroll
    for (int kk = 0; kk < 2 * Dn; kk += 32) {
      v16bf a = frag_a_rowmajor(arow, kk, half);
#pragma unroll
      for (int nt = 0; nt < 8; ++nt)
        acc[nt] = wmma_bf16(a, frag_b_t(&w1t[nt * 16 + (lane & 15)][0], kk, half), acc[nt]);
    }
#pragma unroll
    for (int nt = 0; nt < 8; ++nt) {
      int col = nt * 16 + (lane & 15);
      float bias = b1[col];
#pragma unroll
      for (int i = 0; i < 8; ++i) {
        int row = m0 + i + 8 * half;
        float hv = acc[nt][i] + bias;
        hv = hv + alpha * hv * __expf(-0.5f * hv * hv);
        hbuf[row][col] = (__bf16)hv;
      }
    }
  }
  __syncthreads();

  // GEMM2: ho(128x64) = h(128x128) @ W2
  {
    v8f acc[4] = {};
    const __bf16* arow = &hbuf[m0 + (lane & 15)][0];
#pragma unroll
    for (int kk = 0; kk < 2 * Dn; kk += 32) {
      v16bf a = frag_a_rowmajor(arow, kk, half);
#pragma unroll
      for (int nt = 0; nt < 4; ++nt)
        acc[nt] = wmma_bf16(a, frag_b_t(&w2t[nt * 16 + (lane & 15)][0], kk, half), acc[nt]);
    }
#pragma unroll
    for (int nt = 0; nt < 4; ++nt) {
      int col = nt * 16 + (lane & 15);
      float bias = b2[col];
#pragma unroll
      for (int i = 0; i < 8; ++i) {
        int row = m0 + i + 8 * half;  // token in chunk
        mlp_out[row0 + (size_t)row * Cn + col] = (__bf16)(acc[nt][i] + bias);
      }
    }
  }
}

// ---------------------------------------------------------------- K4: proj GEMM + LN(C) + residual
__global__ __launch_bounds__(256) void k_proj_ln(
    const __bf16* __restrict__ mlp_out, const __bf16* __restrict__ pWt,
    const float* __restrict__ pb, const float* __restrict__ ln_g,
    const float* __restrict__ ln_b, const float* __restrict__ x,
    float* __restrict__ out) {
  __shared__ __attribute__((aligned(16))) __bf16 a_tile[16][Cn];  // 32 KB
  __shared__ float red_s[8][16];
  __shared__ float red_q[8][16];
  __shared__ float mu_s[16];
  __shared__ float rs_s[16];

  const int tid = threadIdx.x, lane = tid & 31, w = tid >> 5, half = lane >> 4;
  const size_t R0 = (size_t)blockIdx.x * 16;

#if HAVE_TDM
  if (w == 0) {
    // 16 full rows of mlp_out are contiguous: flat 16384 bf16 DMA
    tdm_load_2d(mlp_out + R0 * Cn, lds_addr32(&a_tile[0][0]),
                16 * Cn, 1, 16 * Cn, 1, 16 * Cn, 1, 0, 0, 0);
    __builtin_amdgcn_s_wait_tensorcnt(0);
  }
#else
  for (int i = tid; i < 16 * Cn; i += 256)
    (&a_tile[0][0])[i] = mlp_out[(R0 + (i >> 10)) * Cn + (i & 1023)];
#endif
  __syncthreads();

  const int n0 = w * 128;
  v8f acc[8] = {};
  const __bf16* arow = &a_tile[lane & 15][0];
  const __bf16* bbase = pWt + (size_t)(n0 + (lane & 15)) * Cn;
  for (int kk = 0; kk < Cn; kk += 32) {
    v16bf a = frag_a_rowmajor(arow, kk, half);
#pragma unroll
    for (int nt = 0; nt < 8; ++nt)
      acc[nt] = wmma_bf16(a, frag_b_t(bbase + (size_t)nt * 16 * Cn, kk, half), acc[nt]);
  }

  // bias + per-lane partial row stats
  float ps[8], pq[8];
#pragma unroll
  for (int i = 0; i < 8; ++i) { ps[i] = 0.f; pq[i] = 0.f; }
#pragma unroll
  for (int nt = 0; nt < 8; ++nt) {
    int col = n0 + nt * 16 + (lane & 15);
    float bias = pb[col];
#pragma unroll
    for (int i = 0; i < 8; ++i) {
      float v = acc[nt][i] + bias;
      acc[nt][i] = v;
      ps[i] += v;
      pq[i] += v * v;
    }
  }
#pragma unroll
  for (int i = 0; i < 8; ++i) {
#pragma unroll
    for (int m = 8; m > 0; m >>= 1) {
      ps[i] += __shfl_xor(ps[i], m, 32);
      pq[i] += __shfl_xor(pq[i], m, 32);
    }
  }
  if ((lane & 15) == 0) {
#pragma unroll
    for (int i = 0; i < 8; ++i) {
      red_s[w][half * 8 + i] = ps[i];
      red_q[w][half * 8 + i] = pq[i];
    }
  }
  __syncthreads();
  if (tid < 16) {
    float s = 0.f, q = 0.f;
#pragma unroll
    for (int ww = 0; ww < 8; ++ww) { s += red_s[ww][tid]; q += red_q[ww][tid]; }
    float mu = s * (1.0f / 1024.0f);
    float var = q * (1.0f / 1024.0f) - mu * mu;
    mu_s[tid] = mu;
    rs_s[tid] = rsqrtf(var + EPSf);
  }
  __syncthreads();
#pragma unroll
  for (int nt = 0; nt < 8; ++nt) {
    int col = n0 + nt * 16 + (lane & 15);
    float g = ln_g[col], bb = ln_b[col];
#pragma unroll
    for (int i = 0; i < 8; ++i) {
      int row = i + 8 * half;
      size_t R = R0 + row;
      out[R * Cn + col] = (acc[nt][i] - mu_s[row]) * rs_s[row] * g + bb + x[R * Cn + col];
    }
  }
}

// ---------------------------------------------------------------- launch
extern "C" void kernel_launch(void* const* d_in, const int* in_sizes, int n_in,
                              void* d_out, int out_size, void* d_ws, size_t ws_size,
                              hipStream_t stream) {
  const float* x       = (const float*)d_in[0];
  const float* mark_W  = (const float*)d_in[1];
  const float* mark_b  = (const float*)d_in[2];
  const float* gate_W  = (const float*)d_in[3];
  const float* gate_b  = (const float*)d_in[4];
  const float* carry_g = (const float*)d_in[5];
  const float* carry_b = (const float*)d_in[6];
  const float* card_g  = (const float*)d_in[7];
  const float* card_b  = (const float*)d_in[8];
  const float* W1      = (const float*)d_in[9];
  const float* b1      = (const float*)d_in[10];
  const float* alpha   = (const float*)d_in[11];
  const float* W2      = (const float*)d_in[12];
  const float* b2      = (const float*)d_in[13];
  const float* proj_W  = (const float*)d_in[14];
  const float* proj_b  = (const float*)d_in[15];
  const float* ln_g    = (const float*)d_in[16];
  const float* ln_b    = (const float*)d_in[17];
  float* out = (float*)d_out;

  char* ws = (char*)d_ws;
  size_t off = 0;
  auto alloc = [&](size_t bytes) -> void* {
    void* p = ws + off;
    off = (off + bytes + 255) & ~(size_t)255;
    return p;
  };
  __bf16* xb   = (__bf16*)alloc((size_t)Mn * Cn * 2);            // 32 MB
  __bf16* mWt  = (__bf16*)alloc((size_t)Cn * Cn * 2);            // 2 MB
  __bf16* gWt  = (__bf16*)alloc((size_t)Cn * Cn * 2);            // 2 MB
  __bf16* pWt  = (__bf16*)alloc((size_t)Cn * Cn * 2);            // 2 MB
  __bf16* w1t  = (__bf16*)alloc((size_t)128 * 128 * 2);
  __bf16* w2t  = (__bf16*)alloc((size_t)64 * 128 * 2);
  float*  gated= (float*) alloc((size_t)Mn * Cn * 4);            // 64 MB
  float*  csum = (float*) alloc((size_t)Bn * Hn * NCH * Dn * 4); // 512 KB
  float*  ncar = (float*) alloc((size_t)Bn * Hn * NCH * Dn * 4); // 512 KB
  __bf16* mlp  = (__bf16*)alloc((size_t)Mn * Cn * 2);            // 32 MB
  (void)ws_size; (void)in_sizes; (void)n_in; (void)out_size;

  k_cvt_bf16<<<4096, 256, 0, stream>>>(x, xb, Mn * Cn);
  k_transpose_bf16<<<1024, 256, 0, stream>>>(mark_W, mWt, Cn, Cn);
  k_transpose_bf16<<<1024, 256, 0, stream>>>(gate_W, gWt, Cn, Cn);
  k_transpose_bf16<<<1024, 256, 0, stream>>>(proj_W, pWt, Cn, Cn);
  k_transpose_bf16<<<16, 256, 0, stream>>>(W1, w1t, 128, 128);
  k_transpose_bf16<<<16, 256, 0, stream>>>(W2, w2t, 128, 64);

  dim3 g1(Mn / 64, Cn / 64);
  k_gated_gemm<<<g1, 256, 0, stream>>>(xb, mWt, gWt, mark_b, gate_b, gated);
  k_chunk_sums<<<Bn * Hn * NCH, 64, 0, stream>>>(gated, csum);
  k_carry_ln<<<Bn * Hn, 32, 0, stream>>>(csum, carry_g, carry_b, ncar);
  k_scan_mlp<<<Bn * Hn * NCH, 256, 0, stream>>>(gated, xb, ncar, card_g, card_b,
                                                w1t, b1, alpha, w2t, b2, mlp);
  k_proj_ln<<<Mn / 16, 256, 0, stream>>>(mlp, pWt, proj_b, ln_g, ln_b, x, out);
}
